// GRU_8203387535621
// MI455X (gfx1250) — compile-verified, matrix operand order checked by reading, not run on hardware
//
#include <hip/hip_runtime.h>

// ---------------------------------------------------------------------------
// GRU (L=2, B=32, T=1024, H=512) persistent-scan for MI455X / gfx1250.
//
// Latency-bound recurrence -> persistent grid of 32 workgroups (16 per layer,
// software-pipelined with 1-step skew), GRU weights in bf16 pinned in LDS
// (~285KB of the 320KB/WGP), per-step GEMMs via v_wmma_f32_16x16x32_bf16
// (f32 accum), hidden-state staging via the Tensor Data Mover
// (tensor_load_to_lds + s_wait_tensorcnt), cross-WGP sync via a
// sense-reversing global barrier in workspace.
// ---------------------------------------------------------------------------

typedef unsigned short u16;
typedef __attribute__((ext_vector_type(16))) __bf16 v16bf;
typedef __attribute__((ext_vector_type(8)))  float  v8f;
typedef __attribute__((ext_vector_type(8)))  u16    v8u16;
typedef __attribute__((ext_vector_type(4)))  unsigned int v4u;
typedef __attribute__((ext_vector_type(8)))  int    v8i;
typedef __attribute__((ext_vector_type(4)))  int    v4i;

#define L_   2
#define B_   32
#define T_   1024
#define H_   512
#define G3_  1536
#define NWG  32            // 16 WGs for layer 0 + 16 for layer 1
#define NTHREADS 384       // 12 wave32 -> one 16x16 C-tile per wave
#define COLS 32            // hidden columns owned per WG
#define WROWS 96           // 3 gates * COLS weight rows per matrix slice

// ---- LDS layout (bytes); all offsets 16B aligned -------------------------
#define OFF_WIH    0                               // [96][512] bf16 = 98304
#define OFF_WHH    (WROWS * H_ * 2)                // 98304
#define OFF_STAGE1 (2 * WROWS * H_ * 2)            // 196608: A op 1 [32][512] bf16
#define OFF_STAGE2 (OFF_STAGE1 + B_ * H_ * 2)      // 229376: A op 2 (TDM target)
#define OFF_ACCX   (OFF_STAGE2 + B_ * H_ * 2)      // 262144: [3][32][32] f32
#define OFF_ACCH   (OFF_ACCX + 3 * B_ * COLS * 4)  // 274432
#define OFF_HKEEP  (OFF_ACCH + 3 * B_ * COLS * 4)  // 286720: [32][32] f32
#define OFF_BIH    (OFF_HKEEP + B_ * COLS * 4)     // 290816: [96] f32
#define OFF_BHH    (OFF_BIH + WROWS * 4)           // 291200
#define SMEM_BYTES (OFF_BHH + WROWS * 4)           // 291584 bytes < 320KB

union FragU { v16bf bf; v8u16 h[2]; };

__device__ __forceinline__ u16 f2bf(float f) {
    unsigned u = __float_as_uint(f);
    unsigned r = u + 0x7FFFu + ((u >> 16) & 1u);   // round-to-nearest-even
    return (u16)(r >> 16);
}
__device__ __forceinline__ float bf2f(u16 h) {
    return __uint_as_float(((unsigned)h) << 16);
}
__device__ __forceinline__ float sigm(float x) {
    return 1.0f / (1.0f + __expf(-x));
}
__device__ __forceinline__ float tanh_fast(float x) {
    float a = fabsf(x);
    float e = __expf(-2.0f * a);
    float r = (1.0f - e) / (1.0f + e);
    return copysignf(r, x);
}

// Sense-reversing grid barrier (counters in workspace, re-zeroed per launch).
__device__ __forceinline__ void gridBarrier(unsigned* bar) {
    __threadfence();
    __syncthreads();
    if (threadIdx.x == 0) {
        unsigned g = __hip_atomic_load(&bar[1], __ATOMIC_RELAXED, __HIP_MEMORY_SCOPE_AGENT);
        unsigned a = __hip_atomic_fetch_add(&bar[0], 1u, __ATOMIC_ACQ_REL, __HIP_MEMORY_SCOPE_AGENT);
        if (a == NWG - 1u) {
            __hip_atomic_store(&bar[0], 0u, __ATOMIC_RELAXED, __HIP_MEMORY_SCOPE_AGENT);
            __hip_atomic_fetch_add(&bar[1], 1u, __ATOMIC_ACQ_REL, __HIP_MEMORY_SCOPE_AGENT);
        } else {
            while (__hip_atomic_load(&bar[1], __ATOMIC_ACQUIRE, __HIP_MEMORY_SCOPE_AGENT) == g) {
                __builtin_amdgcn_s_sleep(2);
            }
        }
    }
    __syncthreads();
}

// TDM: DMA a contiguous 32KB hidden-state vector (8192 dwords) into LDS.
// D# built per CDNA5 ISA ch.8: 1-row tile, data_size=4B,
// tile_dim0 = tensor_dim0 = tensor_dim0_stride = 8192.
__device__ __forceinline__ void tdm_load_h(unsigned lds_byte_off, const u16* src) {
    const unsigned long long ga = (unsigned long long)src;
    v4u g0;
    g0.x = 1u;                                              // count=1, user desc
    g0.y = (unsigned)__builtin_amdgcn_groupstaticsize() + lds_byte_off; // lds_addr
    g0.z = (unsigned)(ga & 0xFFFFFFFFu);                    // global_addr[31:0]
    g0.w = (unsigned)((ga >> 32) & 0x1FFFFFFu) | (2u << 30);// ga[56:32] | type=2
    v8i g1;
    g1[0] = (int)(2u << 16);          // workgroup_mask=0, data_size=4B
    g1[1] = (int)(8192u << 16);       // tensor_dim0[15:0] in bits[63:48]
    g1[2] = (int)(1u << 16);          // tensor_dim0 hi=0, tensor_dim1=1
    g1[3] = (int)(8192u << 16);       // tile_dim0 = 8192 in bits[127:112]
    g1[4] = 1;                        // tile_dim1 = 1, tile_dim2 = 0
    g1[5] = 8192;                     // tensor_dim0_stride[31:0]
    g1[6] = 0;                        // stride hi, dim1_stride lo
    g1[7] = 0;
    const v4i z4 = {0, 0, 0, 0};
    const v8i z8 = {0, 0, 0, 0, 0, 0, 0, 0};
    __builtin_amdgcn_tensor_load_to_lds(g0, g1, z4, z4, z8, 0);
}

// One GEMM pass: D[32 x 96gate-slice] += A[32x512] * W_slice^T, bf16 WMMA.
// wave w: m = w/6 (batch half), tile n = w%6 -> gate g = n/2, col-half = n&1.
__device__ __forceinline__ void gemm_pass(const u16* __restrict__ stage,
                                          const u16* __restrict__ w,
                                          float* __restrict__ accbuf,
                                          int wave, int lane) {
    const int m    = wave / 6;
    const int nI   = wave % 6;
    const int g    = nI >> 1;
    const int ch   = nI & 1;
    const int lrow = lane & 15;
    const int arow = m * 16 + lrow;                 // batch row of A tile
    const int wrow = g * 32 + ch * 16 + lrow;       // weight row == B column
    const int koff = (lane < 16) ? 0 : 8;           // ISA 16-bit A/B lane split

    v8f acc = {};
    const u16* ap = stage + arow * H_;
    const u16* bp = w + wrow * H_;
#pragma unroll
    for (int k = 0; k < 16; ++k) {
        const int kb = k * 32 + koff;
        FragU A, Bf;
        A.h[0]  = *(const v8u16*)(ap + kb);
        A.h[1]  = *(const v8u16*)(ap + kb + 16);
        Bf.h[0] = *(const v8u16*)(bp + kb);
        Bf.h[1] = *(const v8u16*)(bp + kb + 16);
        acc = __builtin_amdgcn_wmma_f32_16x16x32_bf16(
            false, A.bf, false, Bf.bf, (short)0, acc, false, false);
    }
    // C/D layout: VGPR r, lane<16 -> (M=r, N=lane); lane>=16 -> (M=r+8, N=lane-16)
    const int Mb = m * 16 + ((lane >= 16) ? 8 : 0);
    const int Nb = ch * 16 + (lane & 15);
#pragma unroll
    for (int r = 0; r < 8; ++r)
        accbuf[g * (B_ * COLS) + (Mb + r) * COLS + Nb] = acc[r];
}

#define HB(l, p) (hbuf + (((l) * 2 + (p)) * (B_ * H_)))

__global__ void gru_init_barrier(unsigned* bar) { bar[0] = 0u; bar[1] = 0u; }

__global__ void __launch_bounds__(NTHREADS, 1)
gru_scan(const float* __restrict__ x, const float* __restrict__ h0,
         const float* __restrict__ Wih, const float* __restrict__ bih,
         const float* __restrict__ Whh, const float* __restrict__ bhh,
         u16* __restrict__ hbuf, unsigned* __restrict__ bar) {
    extern __shared__ char smem[];
    u16*   wih    = (u16*)(smem + OFF_WIH);
    u16*   whh    = (u16*)(smem + OFF_WHH);
    u16*   stage1 = (u16*)(smem + OFF_STAGE1);
    u16*   stage2 = (u16*)(smem + OFF_STAGE2);
    float* accX   = (float*)(smem + OFF_ACCX);
    float* accH   = (float*)(smem + OFF_ACCH);
    float* hkeep  = (float*)(smem + OFF_HKEEP);
    float* bihL   = (float*)(smem + OFF_BIH);
    float* bhhL   = (float*)(smem + OFF_BHH);

    const int wg    = blockIdx.x;
    const int layer = wg >> 4;          // 0 or 1
    const int slice = wg & 15;
    const int j0    = slice * COLS;
    const int tid   = threadIdx.x;
    const int wave  = tid >> 5;
    const int lane  = tid & 31;

    // ---- one-time init: pin bf16 weight slices in LDS --------------------
    for (int idx = tid; idx < WROWS * H_; idx += NTHREADS) {
        const int lr = idx >> 9;               // local row: g*32 + j
        const int k  = idx & (H_ - 1);
        const int g  = lr >> 5;
        const int j  = lr & 31;
        const int grow = g * H_ + j0 + j;      // global gate row in [0,1536)
        wih[idx] = f2bf(Wih[(layer * G3_ + grow) * H_ + k]);
        whh[idx] = f2bf(Whh[(layer * G3_ + grow) * H_ + k]);
    }
    if (tid < WROWS) {
        const int g = tid >> 5, j = tid & 31;
        bihL[tid] = bih[layer * G3_ + g * H_ + j0 + j];
        bhhL[tid] = bhh[layer * G3_ + g * H_ + j0 + j];
    }
    // h state: f32 copy of own columns in LDS + bf16 full vector at parity 1
    for (int e = tid; e < B_ * COLS; e += NTHREADS) {
        const int row = e >> 5, col = e & 31;
        const float hv = h0[(layer * B_ + row) * H_ + j0 + col];
        hkeep[e] = hv;
        HB(layer, 1)[row * H_ + j0 + col] = f2bf(hv);
    }
    __syncthreads();
    gridBarrier(bar);

    // ---- pipelined scan: layer0 computes t=i, layer1 computes t=i-1 ------
#pragma unroll 1
    for (int i = 0; i <= T_; ++i) {
        const bool active = (layer == 0) ? (i < T_) : (i >= 1);
        if (active) {
            const int pw = (layer == 0) ? (i & 1) : ((i + 1) & 1); // write parity
            const int pr = 1 - pw;                                 // own prev parity

            // Issue TDM DMAs up-front: both h operands exist at iter start.
            if (wave == 0) {
                if (layer == 1) tdm_load_h(OFF_STAGE1, HB(0, (i + 1) & 1)); // h0[t]
                tdm_load_h(OFF_STAGE2, HB(layer, pr));                      // own h prev
            }

            // layer 0 pass-1 input: f32 x[t] -> bf16 stage1 (VALU convert)
            if (layer == 0) {
                const int t = i;
                for (int idx = tid; idx < B_ * H_; idx += NTHREADS) {
                    const int b = idx >> 9, k = idx & (H_ - 1);
                    stage1[idx] = f2bf(x[(b * T_ + t) * H_ + k]);
                }
                // hint next timestep's activations toward the caches
                if (t + 1 < T_ && tid < 256) {
                    const int b = tid >> 3, c = tid & 7;
                    __builtin_prefetch(x + (b * T_ + t + 1) * H_ + c * 64, 0, 1);
                }
            }
            // layer 1: first DMA (stage1) must be done; TENSORcnt is in-order
            if (wave == 0 && layer == 1) __builtin_amdgcn_s_wait_tensorcnt(1);
            __syncthreads();
            gemm_pass(stage1, wih, accX, wave, lane);

            // second DMA (stage2) must be done before pass 2
            if (wave == 0) __builtin_amdgcn_s_wait_tensorcnt(0);
            __syncthreads();
            gemm_pass(stage2, whh, accH, wave, lane);
            __syncthreads();

            // elementwise GRU update for owned columns
            u16* hw = HB(layer, pw);
            for (int e = tid; e < B_ * COLS; e += NTHREADS) {
                const int row = e >> 5, col = e & 31;
                const float r = sigm(accX[e] + bihL[col] +
                                     accH[e] + bhhL[col]);
                const float u = sigm(accX[1024 + e] + bihL[32 + col] +
                                     accH[1024 + e] + bhhL[32 + col]);
                const float n = tanh_fast(accX[2048 + e] + bihL[64 + col] +
                                          r * (accH[2048 + e] + bhhL[64 + col]));
                const float hp = hkeep[e];
                const float hn = u * hp + (1.0f - u) * n;
                hkeep[e] = hn;
                hw[row * H_ + j0 + col] = f2bf(hn);
            }
        }
        gridBarrier(bar);
    }
}

// out[b][o] = sum_k h1_final[b][k] * fcW[o][k] + fcb[o]
__global__ void __launch_bounds__(256)
gru_fc(const u16* __restrict__ h1, const float* __restrict__ W,
       const float* __restrict__ bvec, float* __restrict__ out) {
    const int o = blockIdx.x * blockDim.x + threadIdx.x;   // [0, 32*512)
    const int b = o >> 9;
    const int oc = o & (H_ - 1);
    const u16* hr = h1 + b * H_;
    const float* wr = W + oc * H_;
    float s = bvec[oc];
#pragma unroll 4
    for (int k = 0; k < H_; ++k) s += bf2f(hr[k]) * wr[k];
    out[b * H_ + oc] = s;
}

extern "C" void kernel_launch(void* const* d_in, const int* in_sizes, int n_in,
                              void* d_out, int out_size, void* d_ws, size_t ws_size,
                              hipStream_t stream) {
    (void)in_sizes; (void)n_in; (void)out_size; (void)ws_size;
    const float* x   = (const float*)d_in[0];
    const float* h0  = (const float*)d_in[1];
    const float* Wih = (const float*)d_in[2];
    const float* bih = (const float*)d_in[3];
    const float* Whh = (const float*)d_in[4];
    const float* bhh = (const float*)d_in[5];
    const float* fcW = (const float*)d_in[6];
    const float* fcb = (const float*)d_in[7];
    float* out = (float*)d_out;

    unsigned* bar = (unsigned*)d_ws;                 // 2 barrier words
    u16* hbuf = (u16*)((char*)d_ws + 1024);          // [L][2][B][H] bf16

    gru_init_barrier<<<dim3(1), dim3(1), 0, stream>>>(bar);
    gru_scan<<<dim3(NWG), dim3(NTHREADS), SMEM_BYTES, stream>>>(
        x, h0, Wih, bih, Whh, bhh, hbuf, bar);
    // final h of top layer (t = T-1, T even -> parity 1): hbuf[layer=1][p=1]
    gru_fc<<<dim3((B_ * H_) / 256), dim3(256), 0, stream>>>(
        hbuf + (1 * 2 + 1) * (B_ * H_), fcW, fcb, out);
}
